// StageA_GNN_90056874262559
// MI455X (gfx1250) — matrix operand order repaired and checked by reading, not compiled, not run on hardware
//
#include <hip/hip_runtime.h>
#include <hip/hip_bf16.h>
#include <math.h>

#define NN 50000
#define CC 1000
#define EE 800000
#define HH 64
#define KK 32
#define RR 16
#define SS 8
#define LL 2

typedef __bf16 bf16_t;
typedef __attribute__((ext_vector_type(16))) __bf16 v16bf;
typedef __attribute__((ext_vector_type(8)))  __bf16 bf16x8;
typedef __attribute__((ext_vector_type(4)))  __bf16 bf16x4;
typedef __attribute__((ext_vector_type(8)))  float  v8f;

__device__ __forceinline__ float sigmoidf_(float x){ return 1.0f/(1.0f+__expf(-x)); }
__device__ __forceinline__ float softplusf_(float x){ return x > 20.0f ? x : log1pf(__expf(x)); }
__device__ __forceinline__ float reluf_(float x){ return x > 0.0f ? x : 0.0f; }

__device__ __forceinline__ v8f zero8(){
  v8f z;
#pragma unroll
  for (int e=0;e<8;++e) z[e]=0.0f;
  return z;
}
__device__ __forceinline__ bf16x4 cvt4(float4 v){
  bf16x4 r; r[0]=(__bf16)v.x; r[1]=(__bf16)v.y; r[2]=(__bf16)v.z; r[3]=(__bf16)v.w; return r;
}
__device__ __forceinline__ float4 zero4(){ float4 z; z.x=z.y=z.z=z.w=0.0f; return z; }

// A fragment (16x32 bf16, row-major tile): two contiguous 8-elem K-runs -> 2x 16B loads
__device__ __forceinline__ v16bf load_frag_a(const bf16_t* lds, int row0, int k0, int ld){
  const int lane = threadIdx.x & 31;
  const bf16_t* p = lds + (row0 + (lane & 15))*ld + k0 + ((lane >> 4)*8);
  bf16x8 lo = *(const bf16x8*)p;
  bf16x8 hi = *(const bf16x8*)(p + 16);
  v16bf r;
#pragma unroll
  for (int e=0;e<8;++e){ r[e]=lo[e]; r[e+8]=hi[e]; }
  return r;
}

// B fragment from a TRANSPOSED [n][k] tile (LDS or global): two 8-elem K-runs (k0+8h, k0+16+8h)
__device__ __forceinline__ v16bf load_frag_bt(const bf16_t* bT, int k0, int n0, int ldk){
  const int lane = threadIdx.x & 31;
  const bf16_t* p = bT + (size_t)(n0 + (lane & 15))*ldk + k0 + ((lane >> 4)*8);
  bf16x8 lo = *(const bf16x8*)p;
  bf16x8 hi = *(const bf16x8*)(p + 16);
  v16bf r;
#pragma unroll
  for (int e=0;e<8;++e){ r[e]=lo[e]; r[e+8]=hi[e]; }
  return r;
}

__device__ __forceinline__ v8f wmma_bf16(v16bf a, v16bf b, v8f c){
  return __builtin_amdgcn_wmma_f32_16x16x32_bf16(false, a, false, b, (short)0, c, false, false);
}

// ---------------------------------------------------------------------------
// prep: rho, Pmean, and ALL transposed bf16 weight copies (done once; every
// later kernel loads B fragments straight from these with global_load_b128).
// ---------------------------------------------------------------------------
__global__ void k_prep(const float* __restrict__ rho_raw, const float* __restrict__ W_raw,
                       const float* __restrict__ P,
                       const float* __restrict__ enc_w1, const float* __restrict__ enc_w2,
                       const float* __restrict__ gate_w1,
                       const float* __restrict__ upd_w1, const float* __restrict__ upd_w2,
                       const float* __restrict__ toU_w,
                       float* __restrict__ rho, float* __restrict__ Pmean,
                       bf16_t* __restrict__ W1t,  bf16_t* __restrict__ W2t,
                       bf16_t* __restrict__ GW1t, bf16_t* __restrict__ U1t,
                       bf16_t* __restrict__ U2t,  bf16_t* __restrict__ TUWt,
                       bf16_t* __restrict__ WspT){
  const int idx = blockIdx.x * 256 + threadIdx.x;
  if (idx < NN) rho[idx] = sigmoidf_(rho_raw[idx]);
  if (idx < RR){
    float s = 0.0f;
    for (int j = 0; j < SS; ++j) s += P[j*RR + idx];
    Pmean[idx] = s / (float)SS;
  }
  if (idx < 64*1024){                       // enc_w1^T, K padded 1000->1024
    const int n = idx >> 10, k = idx & 1023;
    W1t[idx] = (k < CC) ? (bf16_t)enc_w1[k*HH + n] : (bf16_t)0.0f;
  }
  if (idx < 64*64){                         // enc_w2^T
    const int n = idx >> 6, k = idx & 63;
    W2t[idx] = (bf16_t)enc_w2[k*HH + n];
  }
  if (idx < 64*160){                        // gate_w1^T, K padded 129->160
    const int n = idx / 160, k = idx % 160;
    GW1t[idx] = (k < 2*HH + 1) ? (bf16_t)gate_w1[k*64 + n] : (bf16_t)0.0f;
  }
  if (idx < LL*64*64){                      // upd_w1^T / upd_w2^T per layer
    const int l = idx >> 12, r2 = idx & 4095;
    const int n = r2 >> 6, k = r2 & 63;
    U1t[idx] = (bf16_t)upd_w1[l*HH*HH + k*HH + n];
    U2t[idx] = (bf16_t)upd_w2[l*HH*HH + k*HH + n];
  }
  if (idx < 32*64){                         // toU_w^T [n(32)][k(64)]
    const int n = idx >> 6, k = idx & 63;
    TUWt[idx] = (bf16_t)toU_w[k*KK + n];
  }
  if (idx < 1024*32){                       // softplus(W_raw), [c][k], rows padded 1000->1024
    const int c = idx >> 5, k = idx & 31;
    WspT[idx] = (c < CC) ? (bf16_t)softplusf_(W_raw[c*KK + k]) : (bf16_t)0.0f;
  }
}

// ---------------------------------------------------------------------------
// encoder: h = relu(relu(X @ W1 + b1) @ W2 + b2), 64 rows/block, double-buffered
// ---------------------------------------------------------------------------
__global__ __launch_bounds__(128) void k_encoder(const float* __restrict__ X,
    const bf16_t* __restrict__ W1t, const float* __restrict__ B1,
    const bf16_t* __restrict__ W2t, const float* __restrict__ B2,
    float* __restrict__ h){
  __shared__ bf16_t sA[2][64*32];
  __shared__ bf16_t sY[64*64];
  const int tid = threadIdx.x, wave = tid >> 5, lane = tid & 31;
  const int row0 = blockIdx.x * 64;

  auto stageX_full = [&](int k0, int buf){
#pragma unroll
    for (int i = 0; i < 4; ++i){
      const int lin = tid + i*128;
      const int r = lin >> 3, c4 = (lin & 7)*4;
      const int gr = min(row0 + r, NN-1);
      *(bf16x4*)&sA[buf][r*32 + c4] = cvt4(*(const float4*)(X + (size_t)gr*CC + k0 + c4));
    }
  };
  auto stageX_tail = [&](int k0, int buf){
#pragma unroll
    for (int i = 0; i < 4; ++i){
      const int lin = tid + i*128;
      const int r = lin >> 3, c4 = (lin & 7)*4;
      const int gr = min(row0 + r, NN-1);
      float4 v = (k0 + c4 + 4 <= CC) ? *(const float4*)(X + (size_t)gr*CC + k0 + c4) : zero4();
      *(bf16x4*)&sA[buf][r*32 + c4] = cvt4(v);
    }
  };

  v8f acc[4];
#pragma unroll
  for (int t = 0; t < 4; ++t) acc[t] = zero8();

  stageX_full(0, 0);
  __syncthreads();
  const int NKC = 32;
  for (int kc = 0; kc < NKC; ++kc){
    const int cur = kc & 1;
    if (kc < 30){
      stageX_full((kc+1)*32, cur ^ 1);
      __builtin_prefetch(X + (size_t)min(row0 + (tid >> 3), NN-1)*CC + (kc+2)*32, 0, 3);
    } else if (kc == 30){
      stageX_tail(992, cur ^ 1);
    }
    v16bf a = load_frag_a(sA[cur], wave*16, 0, 32);
#pragma unroll
    for (int t = 0; t < 4; ++t){
      v16bf b = load_frag_bt(W1t, kc*32, t*16, 1024);   // global, L2-resident
      acc[t] = wmma_bf16(a, b, acc[t]);
    }
    __syncthreads();
  }

#pragma unroll
  for (int t = 0; t < 4; ++t){
    const int n = t*16 + (lane & 15);
    const float bn = B1[n];
#pragma unroll
    for (int r = 0; r < 8; ++r){
      const int mm = r + 8*(lane >> 4);
      sY[(wave*16 + mm)*64 + n] = (bf16_t)reluf_(acc[t][r] + bn);
    }
  }
  __syncthreads();

  v8f acc2[4];
#pragma unroll
  for (int t = 0; t < 4; ++t) acc2[t] = zero8();
#pragma unroll
  for (int kc = 0; kc < 2; ++kc){
    v16bf a = load_frag_a(sY, wave*16, kc*32, 64);
#pragma unroll
    for (int t = 0; t < 4; ++t){
      v16bf b = load_frag_bt(W2t, kc*32, t*16, 64);
      acc2[t] = wmma_bf16(a, b, acc2[t]);
    }
  }
#pragma unroll
  for (int t = 0; t < 4; ++t){
    const int n = t*16 + (lane & 15);
    const float bn = B2[n];
#pragma unroll
    for (int r = 0; r < 8; ++r){
      const int mm = r + 8*(lane >> 4);
      const int row = row0 + wave*16 + mm;
      if (row < NN) h[row*HH + n] = reluf_(acc2[t][r] + bn);
    }
  }
}

// ---------------------------------------------------------------------------
// edge gate + scatter: 64 edges/block (E is an exact multiple of 64)
// ---------------------------------------------------------------------------
__global__ __launch_bounds__(128) void k_edge(const float* __restrict__ h,
    const int* __restrict__ src, const int* __restrict__ dst,
    const float* __restrict__ base_w, const float* __restrict__ dist,
    const float* __restrict__ rho,
    const bf16_t* __restrict__ GW1t, const float* __restrict__ GB1,
    const float* __restrict__ GW2, const float* __restrict__ GB2,
    float* __restrict__ m_out, float* __restrict__ deg){
  __shared__ bf16_t sA[64*160];    // [h_src | h_dst | dist | pad]
  __shared__ float  gsum[64];
  __shared__ float  wv[64];
  const int tid = threadIdx.x, wave = tid >> 5, lane = tid & 31;
  const int e0 = blockIdx.x * 64;

  {
    const int r = tid >> 1, half = tid & 1;
    const int e = e0 + r;
    if (half == 0){
      const float* hp = h + (size_t)src[e]*HH;
#pragma unroll
      for (int i = 0; i < 16; ++i)
        *(bf16x4*)&sA[r*160 + i*4] = cvt4(*(const float4*)(hp + i*4));
    } else {
      const float* hp = h + (size_t)dst[e]*HH;
#pragma unroll
      for (int i = 0; i < 16; ++i)
        *(bf16x4*)&sA[r*160 + 64 + i*4] = cvt4(*(const float4*)(hp + i*4));
      bf16x4 z; z[0]=(__bf16)dist[e]; z[1]=z[2]=z[3]=(__bf16)0.0f;
      *(bf16x4*)&sA[r*160 + 128] = z;
      bf16x4 zz; zz[0]=zz[1]=zz[2]=zz[3]=(__bf16)0.0f;
#pragma unroll
      for (int i = 0; i < 7; ++i) *(bf16x4*)&sA[r*160 + 132 + i*4] = zz;
    }
  }
  if (tid < 64) gsum[tid] = 0.0f;
  __syncthreads();

  v8f acc[4];
#pragma unroll
  for (int t = 0; t < 4; ++t) acc[t] = zero8();
#pragma unroll
  for (int kc = 0; kc < 5; ++kc){
    v16bf a = load_frag_a(sA, wave*16, kc*32, 160);
#pragma unroll
    for (int t = 0; t < 4; ++t){
      v16bf b = load_frag_bt(GW1t, kc*32, t*16, 160);   // global, L2-resident
      acc[t] = wmma_bf16(a, b, acc[t]);
    }
  }
#pragma unroll
  for (int t = 0; t < 4; ++t){
    const int n = t*16 + (lane & 15);
    const float b1 = GB1[n], g2 = GW2[n];
#pragma unroll
    for (int r = 0; r < 8; ++r){
      const int mm = wave*16 + r + 8*(lane >> 4);
      atomicAdd(&gsum[mm], reluf_(acc[t][r] + b1) * g2);
    }
  }
  __syncthreads();
  if (tid < 64){
    const int e = e0 + tid;
    const float g = sigmoidf_(gsum[tid] + GB2[0]);
    const float w = base_w[e] * g * rho[src[e]] * rho[dst[e]];
    wv[tid] = w;
    atomicAdd(&deg[dst[e]], w);
  }
  __syncthreads();
  {
    const int r = tid >> 1, half = tid & 1;
    const int e = e0 + r;
    const int d = dst[e];
    const float w = wv[r];
#pragma unroll
    for (int i = 0; i < 8; ++i){
      const bf16x4 hv = *(const bf16x4*)&sA[r*160 + half*32 + i*4];
#pragma unroll
      for (int j = 0; j < 4; ++j)
        atomicAdd(&m_out[d*HH + half*32 + i*4 + j], w * (float)hv[j]);
    }
  }
}

// ---------------------------------------------------------------------------
// node update: u = relu(neigh@W1+b1)@W2+b2; h = LN(h+u); 64 rows/block
// ---------------------------------------------------------------------------
__global__ __launch_bounds__(128) void k_update(const float* __restrict__ m_in,
    const float* __restrict__ deg,
    const bf16_t* __restrict__ W1t, const float* __restrict__ B1,
    const bf16_t* __restrict__ W2t, const float* __restrict__ B2,
    const float* __restrict__ lng, const float* __restrict__ lnb,
    float* __restrict__ h){
  __shared__ bf16_t sA[64*64];
  __shared__ bf16_t sY[64*64];
  __shared__ float  sT[64*64];
  const int tid = threadIdx.x, wave = tid >> 5, lane = tid & 31;
  const int row0 = blockIdx.x * 64;

#pragma unroll
  for (int i = 0; i < 8; ++i){
    const int lin = tid + i*128;
    const int r = lin >> 4, c4 = (lin & 15)*4;
    const int row = min(row0 + r, NN-1);
    const float4 hv = *(const float4*)(h + (size_t)row*HH + c4);
    *(float4*)&sT[r*64 + c4] = hv;
    const float4 mv = *(const float4*)(m_in + (size_t)row*HH + c4);
    const float rd = 1.0f/(deg[row] + 1e-8f);
    float4 nv; nv.x=mv.x*rd; nv.y=mv.y*rd; nv.z=mv.z*rd; nv.w=mv.w*rd;
    *(bf16x4*)&sA[r*64 + c4] = cvt4(nv);
  }
  __syncthreads();

  v8f acc[4];
#pragma unroll
  for (int t = 0; t < 4; ++t) acc[t] = zero8();
#pragma unroll
  for (int kc = 0; kc < 2; ++kc){
    v16bf a = load_frag_a(sA, wave*16, kc*32, 64);
#pragma unroll
    for (int t = 0; t < 4; ++t){
      v16bf b = load_frag_bt(W1t, kc*32, t*16, 64);
      acc[t] = wmma_bf16(a, b, acc[t]);
    }
  }
#pragma unroll
  for (int t = 0; t < 4; ++t){
    const int n = t*16 + (lane & 15);
    const float bn = B1[n];
#pragma unroll
    for (int r = 0; r < 8; ++r){
      const int mm = r + 8*(lane >> 4);
      sY[(wave*16 + mm)*64 + n] = (bf16_t)reluf_(acc[t][r] + bn);
    }
  }
  __syncthreads();

  v8f acc2[4];
#pragma unroll
  for (int t = 0; t < 4; ++t) acc2[t] = zero8();
#pragma unroll
  for (int kc = 0; kc < 2; ++kc){
    v16bf a = load_frag_a(sY, wave*16, kc*32, 64);
#pragma unroll
    for (int t = 0; t < 4; ++t){
      v16bf b = load_frag_bt(W2t, kc*32, t*16, 64);
      acc2[t] = wmma_bf16(a, b, acc2[t]);
    }
  }
#pragma unroll
  for (int t = 0; t < 4; ++t){
    const int n = t*16 + (lane & 15);
    const float bn = B2[n];
#pragma unroll
    for (int r = 0; r < 8; ++r){
      const int mm = r + 8*(lane >> 4);
      sT[(wave*16 + mm)*64 + n] += acc2[t][r] + bn;    // h + u
    }
  }
  __syncthreads();
  if (tid < 64){
    const int row = row0 + tid;
    if (row < NN){
      const float4* x4 = (const float4*)&sT[tid*64];
      float mean = 0.0f;
#pragma unroll
      for (int i = 0; i < 16; ++i){ const float4 v = x4[i]; mean += v.x+v.y+v.z+v.w; }
      mean *= (1.0f/64.0f);
      float var = 0.0f;
#pragma unroll
      for (int i = 0; i < 16; ++i){
        const float4 v = x4[i];
        var += (v.x-mean)*(v.x-mean) + (v.y-mean)*(v.y-mean)
             + (v.z-mean)*(v.z-mean) + (v.w-mean)*(v.w-mean);
      }
      var *= (1.0f/64.0f);
      const float inv = rsqrtf(var + 1e-5f);
#pragma unroll
      for (int i = 0; i < 16; ++i){
        const float4 v = x4[i];
        const float4 g = *(const float4*)(lng + i*4);
        const float4 b = *(const float4*)(lnb + i*4);
        float4 o;
        o.x = g.x*(v.x-mean)*inv + b.x;
        o.y = g.y*(v.y-mean)*inv + b.y;
        o.z = g.z*(v.z-mean)*inv + b.z;
        o.w = g.w*(v.w-mean)*inv + b.w;
        *(float4*)(h + (size_t)row*HH + i*4) = o;
      }
    }
  }
}

// ---------------------------------------------------------------------------
// final: U = softplus(h@toU_w+b); mu = exp(clip(log(lib)+log(max(U@W^T,eps))+alpha+bcorr))
// Column loop is barrier-free: W^T/Q/alpha read straight from global (L2-resident).
// ---------------------------------------------------------------------------
__global__ __launch_bounds__(128) void k_final(const float* __restrict__ h,
    const bf16_t* __restrict__ TUWt, const float* __restrict__ TU_B,
    const bf16_t* __restrict__ WspT, const float* __restrict__ alpha,
    const float* __restrict__ P, const float* __restrict__ Pmean,
    const int* __restrict__ sid, const float* __restrict__ lib,
    const float* __restrict__ Q, float* __restrict__ out){
  __shared__ bf16_t sA[64*64];
  __shared__ bf16_t sU[64*32];
  __shared__ float  sP[64*16];
  __shared__ float  sLl[64];
  const int tid = threadIdx.x, wave = tid >> 5, lane = tid & 31;
  const int row0 = blockIdx.x * 64;

#pragma unroll
  for (int i = 0; i < 8; ++i){
    const int lin = tid + i*128;
    const int r = lin >> 4, c4 = (lin & 15)*4;
    const int row = min(row0 + r, NN-1);
    *(bf16x4*)&sA[r*64 + c4] = cvt4(*(const float4*)(h + (size_t)row*HH + c4));
  }
  if (tid < 64){
    const int row = min(row0 + tid, NN-1);
    const int sd = sid[row];
#pragma unroll
    for (int i = 0; i < 4; ++i){
      const float4 p = *(const float4*)(P + sd*RR + i*4);
      const float4 pm = *(const float4*)(Pmean + i*4);
      float4 o; o.x=p.x-pm.x; o.y=p.y-pm.y; o.z=p.z-pm.z; o.w=p.w-pm.w;
      *(float4*)&sP[tid*16 + i*4] = o;
    }
    const float lv = lib[row];
    sLl[tid] = __logf(lv > 1e-8f ? lv : 1e-8f);
  }
  __syncthreads();

  // U = softplus(h @ toU_w + b)
  v8f accu[2];
  accu[0] = zero8(); accu[1] = zero8();
#pragma unroll
  for (int kc = 0; kc < 2; ++kc){
    v16bf a = load_frag_a(sA, wave*16, kc*32, 64);
#pragma unroll
    for (int t = 0; t < 2; ++t){
      v16bf b = load_frag_bt(TUWt, kc*32, t*16, 64);
      accu[t] = wmma_bf16(a, b, accu[t]);
    }
  }
#pragma unroll
  for (int t = 0; t < 2; ++t){
    const int n = t*16 + (lane & 15);
    const float bn = TU_B[n];
#pragma unroll
    for (int r = 0; r < 8; ++r){
      const int mm = r + 8*(lane >> 4);
      sU[(wave*16 + mm)*32 + n] = (bf16_t)softplusf_(accu[t][r] + bn);
    }
  }
  __syncthreads();

  const v16bf aU = load_frag_a(sU, wave*16, 0, 32);   // reused across all col chunks
  const float ll[ ] = { sLl[wave*16 + 0 + 8*(lane>>4)] }; (void)ll;

  for (int ct = 0; ct < 16; ++ct){
    const int c0 = ct * 64;
#pragma unroll
    for (int t = 0; t < 4; ++t){
      v16bf b = load_frag_bt(WspT, 0, c0 + t*16, 32);  // rows padded to 1024
      v8f acc = zero8();
      acc = wmma_bf16(aU, b, acc);                     // K=32 in a single WMMA
      const int n = t*16 + (lane & 15);
      const int c = c0 + n;
      if (c < CC){
        const float av = alpha[c];
        const float4 q0 = *(const float4*)(Q + (size_t)c*RR + 0);
        const float4 q1 = *(const float4*)(Q + (size_t)c*RR + 4);
        const float4 q2 = *(const float4*)(Q + (size_t)c*RR + 8);
        const float4 q3 = *(const float4*)(Q + (size_t)c*RR + 12);
#pragma unroll
        for (int r = 0; r < 8; ++r){
          const int mm = r + 8*(lane >> 4);
          const int row = row0 + wave*16 + mm;
          if (row < NN){
            float dot = acc[r]; if (dot < 1e-8f) dot = 1e-8f;
            const float4* p4 = (const float4*)&sP[(wave*16 + mm)*16];
            const float4 p0 = p4[0], p1 = p4[1], p2 = p4[2], p3 = p4[3];
            float bc = p0.x*q0.x + p0.y*q0.y + p0.z*q0.z + p0.w*q0.w
                     + p1.x*q1.x + p1.y*q1.y + p1.z*q1.z + p1.w*q1.w
                     + p2.x*q2.x + p2.y*q2.y + p2.z*q2.z + p2.w*q2.w
                     + p3.x*q3.x + p3.y*q3.y + p3.z*q3.z + p3.w*q3.w;
            float lm = sLl[wave*16 + mm] + __logf(dot) + av + bc;
            lm = fminf(fmaxf(lm, -20.0f), 20.0f);
            out[(size_t)row*CC + c] = __expf(lm);
          }
        }
      }
    }
  }
}

// ---------------------------------------------------------------------------
extern "C" void kernel_launch(void* const* d_in, const int* in_sizes, int n_in,
                              void* d_out, int out_size, void* d_ws, size_t ws_size,
                              hipStream_t stream){
  (void)in_sizes; (void)n_in; (void)out_size; (void)ws_size;
  const float* x_common = (const float*)d_in[0];
  const int*   src      = (const int*)d_in[1];
  const int*   dst      = (const int*)d_in[2];
  const float* base_w   = (const float*)d_in[3];
  const float* dist     = (const float*)d_in[4];
  const float* lib      = (const float*)d_in[5];
  const int*   sid      = (const int*)d_in[6];
  const float* enc_w1   = (const float*)d_in[7];
  const float* enc_b1   = (const float*)d_in[8];
  const float* enc_w2   = (const float*)d_in[9];
  const float* enc_b2   = (const float*)d_in[10];
  const float* upd_w1   = (const float*)d_in[11];
  const float* upd_b1   = (const float*)d_in[12];
  const float* upd_w2   = (const float*)d_in[13];
  const float* upd_b2   = (const float*)d_in[14];
  const float* ln_g     = (const float*)d_in[15];
  const float* ln_b     = (const float*)d_in[16];
  const float* gate_w1  = (const float*)d_in[17];
  const float* gate_b1  = (const float*)d_in[18];
  const float* gate_w2  = (const float*)d_in[19];
  const float* gate_b2  = (const float*)d_in[20];
  const float* rho_raw  = (const float*)d_in[21];
  const float* toU_w    = (const float*)d_in[22];
  const float* toU_b    = (const float*)d_in[23];
  const float* W_raw    = (const float*)d_in[24];
  const float* alpha    = (const float*)d_in[25];
  const float* P        = (const float*)d_in[26];
  const float* Q        = (const float*)d_in[27];

  float* ws   = (float*)d_ws;
  float* h    = ws;                          // N*H f32
  float* m    = h   + (size_t)NN*HH;         // N*H f32
  float* deg  = m   + (size_t)NN*HH;         // N f32
  float* rho  = deg + NN;                    // N f32
  float* Pm   = rho + NN;                    // R f32
  bf16_t* W1t  = (bf16_t*)(Pm + RR);         // 64*1024
  bf16_t* W2t  = W1t  + 64*1024;             // 64*64
  bf16_t* GW1t = W2t  + 64*64;               // 64*160
  bf16_t* U1t  = GW1t + 64*160;              // L*64*64
  bf16_t* U2t  = U1t  + LL*64*64;            // L*64*64
  bf16_t* TUWt = U2t  + LL*64*64;            // 32*64
  bf16_t* WspT = TUWt + 32*64;               // 1024*32

  k_prep<<<(64*1024 + 255)/256, 256, 0, stream>>>(rho_raw, W_raw, P,
      enc_w1, enc_w2, gate_w1, upd_w1, upd_w2, toU_w,
      rho, Pm, W1t, W2t, GW1t, U1t, U2t, TUWt, WspT);

  const int nb = (NN + 63)/64;
  k_encoder<<<nb, 128, 0, stream>>>(x_common, W1t, enc_b1, W2t, enc_b2, h);

  for (int l = 0; l < LL; ++l){
    hipMemsetAsync(m,   0, (size_t)NN*HH*sizeof(float), stream);
    hipMemsetAsync(deg, 0, (size_t)NN*sizeof(float),    stream);
    k_edge<<<EE/64, 128, 0, stream>>>(h, src, dst, base_w, dist, rho,
        GW1t, gate_b1, gate_w2, gate_b2, m, deg);
    k_update<<<nb, 128, 0, stream>>>(m, deg, U1t + l*HH*HH, upd_b1 + l*HH,
        U2t + l*HH*HH, upd_b2 + l*HH, ln_g + l*HH, ln_b + l*HH, h);
  }

  k_final<<<nb, 128, 0, stream>>>(h, TUWt, toU_b, WspT, alpha, P, Pm, sid, lib, Q,
      (float*)d_out);
}
